// WindowAttention_14285061226731
// MI455X (gfx1250) — matrix operand order repaired and checked by reading, not compiled, not run on hardware
//
#include <hip/hip_runtime.h>
#include <hip/hip_bf16.h>

#define DIMD   384
#define NHEAD  12
#define LWIN   343
#define NWIN   64
#define LPAD   384          // L padded to 24*16 for WMMA tiling
#define HD     32
#define TABLE_SZ 2197
#define M_TOTAL (NWIN * LWIN)   // 21952 = 343*64

typedef __attribute__((ext_vector_type(16))) _Float16 v16h;
typedef __attribute__((ext_vector_type(8)))  _Float16 v8h;
typedef __attribute__((ext_vector_type(8)))  float    v8f;

// ---------------------------------------------------------------------------
// CDNA5 wave32 WMMA fragment loaders (16x16x32 f16, f32 accum).
// A (16xK tile slice, row-major, stride ldk halves):
//   lane: m = lane&15, h = lane>>4
//   halves 0..7  -> K = h*8 + 0..7 ; halves 8..15 -> K = 16 + h*8 + 0..7
// B ([n][k] row-major): lane n = lane&15; same K mapping as A
// C/D f32: c[r] <-> row r + 8*(lane>>4), col lane&15
// ---------------------------------------------------------------------------
__device__ __forceinline__ v16h load_a_frag(const _Float16* p, int ldk) {
  const int lane = threadIdx.x & 31;
  const int m = lane & 15, hh = lane >> 4;
  const _Float16* r = p + m * ldk + hh * 8;
  v8h lo = *(const v8h*)(r);
  v8h hi = *(const v8h*)(r + 16);
  v16h a;
#pragma unroll
  for (int i = 0; i < 8; ++i) { a[i] = lo[i]; a[i + 8] = hi[i]; }
  return a;
}

__device__ __forceinline__ v16h load_b_frag(const _Float16* p, int ldk) {
  const int lane = threadIdx.x & 31;
  const int n = lane & 15, hh = lane >> 4;
  const _Float16* r = p + n * ldk + hh * 8;
  v8h lo = *(const v8h*)(r);
  v8h hi = *(const v8h*)(r + 16);
  v16h b;
#pragma unroll
  for (int i = 0; i < 8; ++i) { b[i] = lo[i]; b[i + 8] = hi[i]; }
  return b;
}

// ---------------------------------------------------------------------------
// CDNA5 async global->LDS copy (ASYNCcnt-tracked), 16 bytes per lane.
// ---------------------------------------------------------------------------
__device__ __forceinline__ void async_copy_b128(_Float16* lds_dst,
                                                const _Float16* gsrc) {
  unsigned ldsoff = (unsigned)(unsigned long long)(void*)lds_dst;
  asm volatile("global_load_async_to_lds_b128 %0, %1, off"
               :: "v"(ldsoff), "v"(gsrc) : "memory");
}
__device__ __forceinline__ void wait_async0() {
  asm volatile("s_wait_asynccnt 0x0" ::: "memory");
}

// ---------------------------------------------------------------------------
// Kernel 1: expand relative-position bias table -> dense [NH, L, L] f32
// ---------------------------------------------------------------------------
__global__ void bias_expand_kernel(const int* __restrict__ rpi,
                                   const float* __restrict__ table,
                                   float* __restrict__ biasF, int total) {
  int i = blockIdx.x * blockDim.x + threadIdx.x;
  if (i >= total) return;
  int h = i / (LWIN * LWIN);
  int idx = i - h * (LWIN * LWIN);
  biasF[i] = table[h * TABLE_SZ + rpi[idx]];
}

// ---------------------------------------------------------------------------
// Kernel 2: QKV GEMM.  C[21952,1152] = x[21952,384] @ w_qkv[1152,384]^T
// 64x64 tile, K-step 64, double-buffered LDS, 8 waves (4M x 2N),
// 4 WMMA per wave per barrier. Stores f16 Q (pre-scaled), K, V per-head.
// ---------------------------------------------------------------------------
__global__ __launch_bounds__(256) void qkv_gemm_kernel(
    const float* __restrict__ x, const float* __restrict__ w_qkv,
    _Float16* __restrict__ q16, _Float16* __restrict__ k16,
    _Float16* __restrict__ v16o) {
  __shared__ __align__(16) _Float16 As[2][64 * 64];
  __shared__ __align__(16) _Float16 Bs[2][64 * 64];

  const int tid = threadIdx.x;
  const int wave = tid >> 5, lane = tid & 31;
  const int wm = wave >> 1, wn = wave & 1;
  const int mbase = blockIdx.x * 64;
  const int nbase = blockIdx.y * 64;
  const int lrow = tid >> 2;            // 0..63
  const int lk = (tid & 3) * 16;        // 0,16,32,48

  v8f acc0 = {}; v8f acc1 = {};
  float ra[16], rb[16];

  {  // prologue: tile kb=0 -> buffer 0
    const float* sA = x + (size_t)(mbase + lrow) * DIMD + lk;
    const float* sB = w_qkv + (size_t)(nbase + lrow) * DIMD + lk;
#pragma unroll
    for (int j = 0; j < 16; ++j) { ra[j] = sA[j]; rb[j] = sB[j]; }
#pragma unroll
    for (int j = 0; j < 16; ++j) {
      As[0][lrow * 64 + lk + j] = (_Float16)ra[j];
      Bs[0][lrow * 64 + lk + j] = (_Float16)rb[j];
    }
  }
  __syncthreads();

  int buf = 0;
  for (int kb = 0; kb < DIMD; kb += 64) {
    const bool has_next = (kb + 64) < DIMD;
    if (has_next) {
      const float* sA = x + (size_t)(mbase + lrow) * DIMD + kb + 64 + lk;
      const float* sB = w_qkv + (size_t)(nbase + lrow) * DIMD + kb + 64 + lk;
#pragma unroll
      for (int j = 0; j < 16; ++j) { ra[j] = sA[j]; rb[j] = sB[j]; }
    }
    const _Float16* At = As[buf] + wm * 16 * 64;
    const _Float16* Bt = Bs[buf] + wn * 32 * 64;
#pragma unroll
    for (int kk = 0; kk < 64; kk += 32) {
      v16h a  = load_a_frag(At + kk, 64);
      v16h b0 = load_b_frag(Bt + kk, 64);
      v16h b1 = load_b_frag(Bt + 16 * 64 + kk, 64);
      acc0 = __builtin_amdgcn_wmma_f32_16x16x32_f16(false, a, false, b0,
                                                    (short)0, acc0, false, false);
      acc1 = __builtin_amdgcn_wmma_f32_16x16x32_f16(false, a, false, b1,
                                                    (short)0, acc1, false, false);
    }
    if (has_next) {
#pragma unroll
      for (int j = 0; j < 16; ++j) {
        As[buf ^ 1][lrow * 64 + lk + j] = (_Float16)ra[j];
        Bs[buf ^ 1][lrow * 64 + lk + j] = (_Float16)rb[j];
      }
    }
    __syncthreads();
    buf ^= 1;
  }

  const float qscale = 0.17677669529663687f;  // 32^-0.5
#pragma unroll
  for (int fc = 0; fc < 2; ++fc) {
    v8f c = fc ? acc1 : acc0;
    int ng = nbase + wn * 32 + fc * 16 + (lane & 15);
    int which = ng / DIMD;
    int rem = ng - which * DIMD;
    int head = rem >> 5;
    int d = rem & 31;
    _Float16* dst = (which == 0) ? q16 : ((which == 1) ? k16 : v16o);
    float sc = (which == 0) ? qscale : 1.0f;
#pragma unroll
    for (int r = 0; r < 8; ++r) {
      int mg = mbase + wm * 16 + r + 8 * (lane >> 4);
      int nwi = mg / LWIN;
      int l = mg - nwi * LWIN;
      dst[(((size_t)nwi * NHEAD + head) * LWIN + l) * HD + d] =
          (_Float16)(c[r] * sc);
    }
  }
}

// ---------------------------------------------------------------------------
// Kernel 3: attention core. One block per (head, window); 256 threads.
// K[LPAD][HD] (async-copied) and V^T[HD][LPAD] resident in LDS f16.
// 64-row Q chunks (async-copied): QK^T (12 frags/wave, 8 waves), bias+mask,
// softmax (shfl width-16 + LDS cross-wave), P->LDS f16, PV (1 frag/wave,
// 12 WMMA), store f16 merged-head output.
// ---------------------------------------------------------------------------
__global__ __launch_bounds__(256) void attn_kernel(
    const _Float16* __restrict__ q16, const _Float16* __restrict__ k16,
    const _Float16* __restrict__ v16i, const float* __restrict__ biasF,
    const unsigned char* __restrict__ mask, _Float16* __restrict__ attn16) {
  __shared__ __align__(16) _Float16 Ksm[LPAD * HD];   // [m][d]
  __shared__ __align__(16) _Float16 Vtsm[HD * LPAD];  // [d][m]
  __shared__ __align__(16) _Float16 Qsm[64 * HD];     // [row][d]
  __shared__ __align__(16) _Float16 Psm[64 * LPAD];   // [row][m]
  __shared__ float redbuf[2][4][2][8];                // [wc][mf][hh][r]

  const int h = blockIdx.x;
  const int n = blockIdx.y;
  const int tid = threadIdx.x;
  const int wave = tid >> 5, lane = tid & 31;
  const int hh = lane >> 4, ln = lane & 15;
  const size_t kvbase = ((size_t)n * NHEAD + h) * LWIN * HD;
  const float* biasH = biasF + (size_t)h * LWIN * LWIN;
  const unsigned char* maskN = mask + (size_t)n * LWIN * LWIN;

  // K: async f16 copy in 16B chunks; pad rows zero-filled.
  for (int c = tid; c < (LPAD * HD) / 8; c += 256) {
    int m = c >> 2;
    int d0 = (c & 3) * 8;
    if (m < LWIN) {
      async_copy_b128(Ksm + c * 8,
                      (const _Float16*)(k16 + kvbase + (size_t)m * HD + d0));
    } else {
      v8h z = {};
      *(v8h*)(Ksm + c * 8) = z;
    }
  }
  // V^T: vector global load, scattered transpose store to LDS.
  for (int c = tid; c < (LPAD * HD) / 8; c += 256) {
    int m = c >> 2;
    int d0 = (c & 3) * 8;
    if (m < LWIN) {
      v8h vv = *(const v8h*)(v16i + kvbase + (size_t)m * HD + d0);
#pragma unroll
      for (int j = 0; j < 8; ++j) Vtsm[(d0 + j) * LPAD + m] = vv[j];
    } else {
#pragma unroll
      for (int j = 0; j < 8; ++j) Vtsm[(d0 + j) * LPAD + m] = (_Float16)0.f;
    }
  }
  wait_async0();
  __syncthreads();

  const int mf = wave >> 1;       // 0..3: 16-row slice within 64-row chunk
  const int wc = wave & 1;        // 0..1: 192-wide column group
  const int nwcol = wc * 192;

  for (int l0 = 0; l0 < LWIN; l0 += 64) {
    {  // Q chunk: async f16 copy, one 16B chunk per thread
      int r = tid >> 2;
      int d0 = (tid & 3) * 8;
      int l = l0 + r;
      if (l < LWIN) {
        async_copy_b128(Qsm + tid * 8,
                        (const _Float16*)(q16 + kvbase + (size_t)l * HD + d0));
      } else {
        v8h z = {};
        *(v8h*)(Qsm + tid * 8) = z;
      }
    }
    wait_async0();
    __syncthreads();

    // ---- S = Q K^T : 12 x 16-col fragments per wave ----
    float s[12][8];
    v16h a = load_a_frag(Qsm + mf * 16 * HD, HD);
#pragma unroll
    for (int f = 0; f < 12; ++f) {
      v16h b = load_b_frag(Ksm + (nwcol + f * 16) * HD, HD);
      v8f c = {};
      c = __builtin_amdgcn_wmma_f32_16x16x32_f16(false, a, false, b,
                                                 (short)0, c, false, false);
#pragma unroll
      for (int r = 0; r < 8; ++r) s[f][r] = c[r];
    }

    // ---- bias + mask (q already carries the 32^-0.5 scale) ----
#pragma unroll
    for (int f = 0; f < 12; ++f) {
      int m = nwcol + f * 16 + ln;
#pragma unroll
      for (int r = 0; r < 8; ++r) {
        int l = l0 + mf * 16 + r + 8 * hh;
        float v = -3.0e38f;
        if (l < LWIN && m < LWIN) {
          int ii = l * LWIN + m;
          if (maskN[ii]) v = s[f][r] + biasH[ii];
        }
        s[f][r] = v;
      }
    }

    // ---- row max: shuffle width-16 + 2-wave LDS combine ----
    float mx[8];
#pragma unroll
    for (int r = 0; r < 8; ++r) {
      float v = s[0][r];
#pragma unroll
      for (int f = 1; f < 12; ++f) v = fmaxf(v, s[f][r]);
      mx[r] = v;
    }
#pragma unroll
    for (int off = 1; off < 16; off <<= 1)
#pragma unroll
      for (int r = 0; r < 8; ++r) mx[r] = fmaxf(mx[r], __shfl_xor(mx[r], off));
    if (ln == 0) {
#pragma unroll
      for (int r = 0; r < 8; ++r) redbuf[wc][mf][hh][r] = mx[r];
    }
    __syncthreads();
#pragma unroll
    for (int r = 0; r < 8; ++r)
      mx[r] = fmaxf(redbuf[0][mf][hh][r], redbuf[1][mf][hh][r]);

    // ---- exp + row sum ----
    float sm[8];
#pragma unroll
    for (int r = 0; r < 8; ++r) sm[r] = 0.f;
#pragma unroll
    for (int f = 0; f < 12; ++f)
#pragma unroll
      for (int r = 0; r < 8; ++r) {
        float e = __expf(s[f][r] - mx[r]);
        s[f][r] = e;
        sm[r] += e;
      }
#pragma unroll
    for (int off = 1; off < 16; off <<= 1)
#pragma unroll
      for (int r = 0; r < 8; ++r) sm[r] += __shfl_xor(sm[r], off);
    __syncthreads();  // all reads of redbuf(max) done before reuse
    if (ln == 0) {
#pragma unroll
      for (int r = 0; r < 8; ++r) redbuf[wc][mf][hh][r] = sm[r];
    }
    __syncthreads();
    float inv[8];
#pragma unroll
    for (int r = 0; r < 8; ++r)
      inv[r] = 1.0f / (redbuf[0][mf][hh][r] + redbuf[1][mf][hh][r]);

    // ---- normalized P -> LDS f16 ----
#pragma unroll
    for (int f = 0; f < 12; ++f) {
      int col = nwcol + f * 16 + ln;
#pragma unroll
      for (int r = 0; r < 8; ++r) {
        int row = mf * 16 + r + 8 * hh;
        Psm[row * LPAD + col] = (_Float16)(s[f][r] * inv[r]);
      }
    }
    __syncthreads();

    // ---- O = P V : 8 waves, one 16x16 frag each (4M x 2N), K=LPAD ----
    {
      const int omf = wave >> 1, onf = wave & 1;
      v8f oc = {};
#pragma unroll
      for (int ks = 0; ks < LPAD; ks += 32) {
        v16h pa = load_a_frag(Psm + omf * 16 * LPAD + ks, LPAD);
        v16h vb = load_b_frag(Vtsm + (onf * 16) * LPAD + ks, LPAD);
        oc = __builtin_amdgcn_wmma_f32_16x16x32_f16(false, pa, false, vb,
                                                    (short)0, oc, false, false);
      }
      int d = onf * 16 + ln;
#pragma unroll
      for (int r = 0; r < 8; ++r) {
        int l = l0 + omf * 16 + r + 8 * hh;
        if (l < LWIN)
          attn16[((size_t)n * LWIN + l) * DIMD + h * HD + d] = (_Float16)oc[r];
      }
    }
    __syncthreads();
  }
}

// ---------------------------------------------------------------------------
// Kernel 4: output projection. out = A16 @ w_proj^T + b_proj (f32 out).
// A tile async-copied global->LDS (already f16); B staged via regs + cvt.
// Double-buffered, K-step 64.
// ---------------------------------------------------------------------------
__global__ __launch_bounds__(256) void proj_gemm_kernel(
    const _Float16* __restrict__ a16, const float* __restrict__ w_proj,
    const float* __restrict__ b_proj, float* __restrict__ out) {
  __shared__ __align__(16) _Float16 As[2][64 * 64];
  __shared__ __align__(16) _Float16 Bs[2][64 * 64];

  const int tid = threadIdx.x;
  const int wave = tid >> 5, lane = tid & 31;
  const int wm = wave >> 1, wn = wave & 1;
  const int mbase = blockIdx.x * 64;
  const int nbase = blockIdx.y * 64;
  const int lrow = tid >> 2;
  const int lk = (tid & 3) * 16;
  const int i0 = lrow * 64 + lk;

  v8f acc0 = {}; v8f acc1 = {};
  float rb[16];

  {  // prologue
    const _Float16* sA = a16 + (size_t)(mbase + lrow) * DIMD + lk;
    async_copy_b128(&As[0][i0], sA);
    async_copy_b128(&As[0][i0 + 8], sA + 8);
    const float* sB = w_proj + (size_t)(nbase + lrow) * DIMD + lk;
#pragma unroll
    for (int j = 0; j < 16; ++j) rb[j] = sB[j];
#pragma unroll
    for (int j = 0; j < 16; ++j) Bs[0][i0 + j] = (_Float16)rb[j];
  }
  wait_async0();
  __syncthreads();

  int buf = 0;
  for (int kb = 0; kb < DIMD; kb += 64) {
    const bool has_next = (kb + 64) < DIMD;
    if (has_next) {
      const _Float16* sA = a16 + (size_t)(mbase + lrow) * DIMD + kb + 64 + lk;
      async_copy_b128(&As[buf ^ 1][i0], sA);
      async_copy_b128(&As[buf ^ 1][i0 + 8], sA + 8);
      const float* sB = w_proj + (size_t)(nbase + lrow) * DIMD + kb + 64 + lk;
#pragma unroll
      for (int j = 0; j < 16; ++j) rb[j] = sB[j];
    }
    const _Float16* At = As[buf] + wm * 16 * 64;
    const _Float16* Bt = Bs[buf] + wn * 32 * 64;
#pragma unroll
    for (int kk = 0; kk < 64; kk += 32) {
      v16h a  = load_a_frag(At + kk, 64);
      v16h b0 = load_b_frag(Bt + kk, 64);
      v16h b1 = load_b_frag(Bt + 16 * 64 + kk, 64);
      acc0 = __builtin_amdgcn_wmma_f32_16x16x32_f16(false, a, false, b0,
                                                    (short)0, acc0, false, false);
      acc1 = __builtin_amdgcn_wmma_f32_16x16x32_f16(false, a, false, b1,
                                                    (short)0, acc1, false, false);
    }
    if (has_next) {
#pragma unroll
      for (int j = 0; j < 16; ++j) Bs[buf ^ 1][i0 + j] = (_Float16)rb[j];
    }
    wait_async0();
    __syncthreads();
    buf ^= 1;
  }

#pragma unroll
  for (int fc = 0; fc < 2; ++fc) {
    v8f c = fc ? acc1 : acc0;
    int ng = nbase + wn * 32 + fc * 16 + (lane & 15);
    float bias = b_proj[ng];
#pragma unroll
    for (int r = 0; r < 8; ++r) {
      int mg = mbase + wm * 16 + r + 8 * (lane >> 4);
      out[(size_t)mg * DIMD + ng] = c[r] + bias;
    }
  }
}

// ---------------------------------------------------------------------------
extern "C" void kernel_launch(void* const* d_in, const int* in_sizes, int n_in,
                              void* d_out, int out_size, void* d_ws,
                              size_t ws_size, hipStream_t stream) {
  const float* x            = (const float*)d_in[0];
  const unsigned char* mask = (const unsigned char*)d_in[1];
  const int* rpi            = (const int*)d_in[2];
  const float* table        = (const float*)d_in[3];
  const float* w_qkv        = (const float*)d_in[4];
  const float* w_proj       = (const float*)d_in[5];
  const float* b_proj       = (const float*)d_in[6];
  float* out                = (float*)d_out;

  char* ws = (char*)d_ws;
  const size_t half_elems = (size_t)M_TOTAL * DIMD;  // 8,429,568
  _Float16* q16    = (_Float16*)ws; ws += half_elems * sizeof(_Float16);
  _Float16* k16    = (_Float16*)ws; ws += half_elems * sizeof(_Float16);
  _Float16* v16b   = (_Float16*)ws; ws += half_elems * sizeof(_Float16);
  _Float16* attn16 = (_Float16*)ws; ws += half_elems * sizeof(_Float16);
  float* biasF     = (float*)ws;

  const int total_bias = NHEAD * LWIN * LWIN;
  bias_expand_kernel<<<(total_bias + 255) / 256, 256, 0, stream>>>(
      rpi, table, biasF, total_bias);
  qkv_gemm_kernel<<<dim3(M_TOTAL / 64, (3 * DIMD) / 64), 256, 0, stream>>>(
      x, w_qkv, q16, k16, v16b);
  attn_kernel<<<dim3(NHEAD, NWIN), 256, 0, stream>>>(q16, k16, v16b, biasF,
                                                     mask, attn16);
  proj_gemm_kernel<<<dim3(M_TOTAL / 64, DIMD / 64), 256, 0, stream>>>(
      attn16, w_proj, b_proj, out);
}
